// RNN_24111946399935
// MI455X (gfx1250) — matrix operand order repaired and checked by reading, not compiled, not run on hardware
//
#include <hip/hip_runtime.h>
#include <cmath>

// ---------------------------------------------------------------------------
// tanh-RNN: out[t,b,h] = tanh(x[b,t,:]@Wx + bx + h_{t-1}@Wh + bh)
// B=128, L=1024, I=256, H=512
//
// Phase 1: big GEMM x@Wx+bx written straight into d_out in [L,B,H] layout.
// Phase 2: persistent 16-block kernel, 1024 steps. Wh slice staged in LDS
//          (ds_load_b128 B-fragments), pipelined global A-fragments, dual
//          WMMA accumulator chains, hw tanh, NT output stores, grid barrier
//          via agent-scope atomics in d_ws.
// ---------------------------------------------------------------------------

typedef __bf16 bf16;
typedef __attribute__((ext_vector_type(16))) __bf16 v16bf;
typedef __attribute__((ext_vector_type(8)))  __bf16 v8bf;
typedef __attribute__((ext_vector_type(8)))  float  v8f;

#define BATCH 128
#define SEQL  1024
#define IDIM  256
#define HDIM  512
#define NBLK_RNN 16

// workspace layout (bytes)
#define WS_WXT   0                         // bf16 WxT [512][256]  = 262144 B
#define WS_WHT   (262144)                  // bf16 WhT [512][512]  = 524288 B
#define WS_HBUF  (262144 + 524288)         // bf16 h   [2][128][512] = 262144 B
#define WS_BAR   (262144 + 524288 + 262144)

// Build a 16-wide bf16 fragment from two 16-byte chunks (two b128 loads).
__device__ __forceinline__ v16bf load_frag16(const bf16* __restrict__ p0,
                                             const bf16* __restrict__ p1) {
  v8bf lo = *(const v8bf*)p0;
  v8bf hi = *(const v8bf*)p1;
  v16bf r;
#pragma unroll
  for (int i = 0; i < 8; ++i) { r[i] = lo[i]; r[i + 8] = hi[i]; }
  return r;
}

// Same, but from LDS (emits ds_load_b128 pairs).
__device__ __forceinline__ v16bf lds_frag16(const bf16* p0, const bf16* p1) {
  v8bf lo = *(const v8bf*)p0;
  v8bf hi = *(const v8bf*)p1;
  v16bf r;
#pragma unroll
  for (int i = 0; i < 8; ++i) { r[i] = lo[i]; r[i + 8] = hi[i]; }
  return r;
}

// gfx1250 has a TANH transcendental; use it when the builtin exists.
__device__ __forceinline__ float hw_tanh(float x) {
#if __has_builtin(__builtin_amdgcn_tanhf)
  return __builtin_amdgcn_tanhf(x);
#elif __has_builtin(__builtin_amdgcn_tanh_f32)
  return __builtin_amdgcn_tanh_f32(x);
#else
  return tanhf(x);
#endif
}

// ---------------------------------------------------------------------------
// Prep: convert + transpose weights to bf16, convert h0 to bf16, reset barrier
// ---------------------------------------------------------------------------
__global__ __launch_bounds__(256) void rnn_prep_kernel(
    const float* __restrict__ Wx, const float* __restrict__ Wh,
    const float* __restrict__ h0, bf16* __restrict__ wxT,
    bf16* __restrict__ whT, bf16* __restrict__ hbuf, int* __restrict__ bar) {
  int i = blockIdx.x * blockDim.x + threadIdx.x;   // up to 512*512
  if (i < IDIM * HDIM) {                           // Wx [256][512] -> WxT [512][256]
    int r = i / HDIM, c = i % HDIM;
    wxT[c * IDIM + r] = (bf16)Wx[i];
  }
  if (i < HDIM * HDIM) {                           // Wh [512][512] -> WhT [512][512]
    int r = i / HDIM, c = i % HDIM;
    whT[c * HDIM + r] = (bf16)Wh[i];
  }
  if (i < BATCH * HDIM) {                          // h0 -> bf16 buffer 0
    hbuf[i] = (bf16)h0[i];
  }
  if (i == 0) *bar = 0;
}

// ---------------------------------------------------------------------------
// Phase 1: out[t*B + b, n] = sum_k x[b, t, k] * Wx[k, n] + bx[n]
// grid: (L*B)/16 = 8192 blocks, 256 threads (8 waves).
// Each block: one 16-row tile x all 512 columns; wave w owns 4 column tiles.
// Bandwidth-bound (~134 MB in, ~268 MB out); occupancy hides L2 latency.
// ---------------------------------------------------------------------------
__global__ __launch_bounds__(256) void rnn_xproj_kernel(
    const float* __restrict__ x, const bf16* __restrict__ wxT,
    const float* __restrict__ bx, float* __restrict__ out) {
  const int lane = threadIdx.x & 31;
  const int wave = threadIdx.x >> 5;     // 0..7
  const int lm = lane & 15;              // row/col within tile
  const int lh = lane >> 4;              // half select
  const int gr0 = blockIdx.x * 16;       // row tile base in (t*B + b) space
  const int t  = gr0 >> 7;               // / 128
  const int b0 = gr0 & 127;
  const int n0 = wave * 64;

  // this lane's A-matrix row in x: row (b0+lm), timestep t
  const float* xrow = x + ((size_t)(b0 + lm) * SEQL + t) * IDIM;

  v8f acc[4] = {v8f{}, v8f{}, v8f{}, v8f{}};

#pragma unroll 2
  for (int kb = 0; kb < IDIM; kb += 32) {
    // A fragment: K = kb + lh*8 + {0..7}, kb + lh*8 + {16..23}, fp32 -> bf16
    const float* pa = xrow + kb + lh * 8;
    float4 q0 = ((const float4*)pa)[0];
    float4 q1 = ((const float4*)pa)[1];
    float4 q2 = ((const float4*)(pa + 16))[0];
    float4 q3 = ((const float4*)(pa + 16))[1];
    v16bf a;
    a[0]=(bf16)q0.x;  a[1]=(bf16)q0.y;  a[2]=(bf16)q0.z;  a[3]=(bf16)q0.w;
    a[4]=(bf16)q1.x;  a[5]=(bf16)q1.y;  a[6]=(bf16)q1.z;  a[7]=(bf16)q1.w;
    a[8]=(bf16)q2.x;  a[9]=(bf16)q2.y;  a[10]=(bf16)q2.z; a[11]=(bf16)q2.w;
    a[12]=(bf16)q3.x; a[13]=(bf16)q3.y; a[14]=(bf16)q3.z; a[15]=(bf16)q3.w;

#pragma unroll
    for (int j = 0; j < 4; ++j) {
      // B fragment from WxT (column-major Wx): column n0+j*16+lm, K contiguous
      const bf16* pb = wxT + (size_t)(n0 + j * 16 + lm) * IDIM + kb + lh * 8;
      v16bf b = load_frag16(pb, pb + 16);
      acc[j] = __builtin_amdgcn_wmma_f32_16x16x32_bf16(
          false, a, false, b, (short)0, acc[j], false, false);
    }
  }

#pragma unroll
  for (int j = 0; j < 4; ++j) {
    const int n = n0 + j * 16 + lm;
    const float bias = bx[n];
#pragma unroll
    for (int r = 0; r < 8; ++r) {
      const int row = gr0 + lh * 8 + r;      // = t*128 + b
      out[(size_t)row * HDIM + n] = acc[j][r] + bias;
    }
  }
}

// ---------------------------------------------------------------------------
// Phase 2: persistent recurrence. 16 blocks x 512 threads (16 waves).
// Block bk owns output columns [bk*32, bk*32+32) -> its WhT slice is a
// CONTIGUOUS 32 KB chunk, staged once into LDS. Wave w owns one 16x16 tile:
// rows (w&7)*16.., local cols (w>>3)*16..
// Per step: acc = h @ Wh (K=512, 16 WMMAs split into two independent chains;
// B from LDS, A pipelined from L2), h' = tanh(acc + xp + bh); xp read from /
// h' NT-stored to d_out[t], bf16 h' to a global double buffer, grid barrier.
// ---------------------------------------------------------------------------
__global__ __launch_bounds__(512) void rnn_scan_kernel(
    const bf16* __restrict__ whT, const float* __restrict__ bh,
    bf16* __restrict__ hbuf, float* __restrict__ out, int* __restrict__ bar) {
  __shared__ bf16 whLDS[32 * HDIM];      // 32 columns x 512 K = 32 KB

  const int lane = threadIdx.x & 31;
  const int wave = threadIdx.x >> 5;     // 0..15
  const int lm = lane & 15;
  const int lh = lane >> 4;
  const int m0 = (wave & 7) * 16;
  const int nloc0 = (wave >> 3) * 16;               // local column-tile (0|16)
  const int n = blockIdx.x * 32 + nloc0 + lm;       // global output column
  const float bias = bh[n];

  // ---- stage this block's WhT slice into LDS (contiguous 32 KB) ----
  {
    const v8bf* src = (const v8bf*)(whT + (size_t)blockIdx.x * 32 * HDIM);
    v8bf* dst = (v8bf*)whLDS;
#pragma unroll
    for (int i = 0; i < (32 * HDIM / 8) / 512; ++i)
      dst[threadIdx.x + i * 512] = src[threadIdx.x + i * 512];
  }
  __syncthreads();

  bf16* buf0 = hbuf;
  bf16* buf1 = hbuf + (size_t)BATCH * HDIM;

  // B fragments: column n (local) of Wh, contiguous K in LDS
  const bf16* bcol = whLDS + (size_t)(nloc0 + lm) * HDIM + lh * 8;

  for (int t = 0; t < SEQL; ++t) {
    const bf16* hc = (t & 1) ? buf1 : buf0;
    bf16*       hn = (t & 1) ? buf0 : buf1;
    const bf16* arow = hc + (size_t)(m0 + lm) * HDIM + lh * 8;
    float* orow = out + (size_t)t * (BATCH * HDIM);

    // hoist epilogue xp loads so they overlap the whole K loop
    float xp[8];
#pragma unroll
    for (int r = 0; r < 8; ++r)
      xp[r] = orow[(size_t)(m0 + lh * 8 + r) * HDIM + n];

    // two independent accumulation chains (K halves) for better XDL overlap
    v8f acc0 = {}, acc1 = {};
    v16bf acur = load_frag16(arow, arow + 16);
#pragma unroll
    for (int kb = 0; kb < HDIM; kb += 32) {
      v16bf anext = {};
      if (kb + 32 < HDIM)                  // prefetch next A fragment
        anext = load_frag16(arow + kb + 32, arow + kb + 48);
      v16bf b = lds_frag16(bcol + kb, bcol + kb + 16);   // ds_load_b128 x2
      if (kb < HDIM / 2)
        acc0 = __builtin_amdgcn_wmma_f32_16x16x32_bf16(
            false, acur, false, b, (short)0, acc0, false, false);
      else
        acc1 = __builtin_amdgcn_wmma_f32_16x16x32_bf16(
            false, acur, false, b, (short)0, acc1, false, false);
      acur = anext;
    }

#pragma unroll
    for (int r = 0; r < 8; ++r) {
      const int m = m0 + lh * 8 + r;
      const size_t idx = (size_t)m * HDIM + n;
      float v = hw_tanh(acc0[r] + acc1[r] + xp[r] + bias);
      __builtin_nontemporal_store(v, &orow[idx]);  // final output, never re-read
      hn[idx] = (bf16)v;             // bf16 copy for next step's A operand
    }

    // ---- grid-wide barrier (16 co-resident blocks) ----
    __syncthreads();
    if (threadIdx.x == 0) {
      __threadfence();  // publish hn stores at agent scope
      __hip_atomic_fetch_add(bar, 1, __ATOMIC_RELEASE, __HIP_MEMORY_SCOPE_AGENT);
      const int target = (t + 1) * NBLK_RNN;
      while (__hip_atomic_load(bar, __ATOMIC_ACQUIRE,
                               __HIP_MEMORY_SCOPE_AGENT) < target) {
        __builtin_amdgcn_s_sleep(1);
      }
      __threadfence();
    }
    __syncthreads();
  }
}

// ---------------------------------------------------------------------------
extern "C" void kernel_launch(void* const* d_in, const int* in_sizes, int n_in,
                              void* d_out, int out_size, void* d_ws, size_t ws_size,
                              hipStream_t stream) {
  const float* x  = (const float*)d_in[0];  // [128,1024,256]
  const float* h0 = (const float*)d_in[1];  // [128,512]
  const float* Wx = (const float*)d_in[2];  // [256,512]
  const float* bx = (const float*)d_in[3];  // [512]
  const float* Wh = (const float*)d_in[4];  // [512,512]
  const float* bh = (const float*)d_in[5];  // [512]
  float* out = (float*)d_out;               // [1024,128,512]

  char* ws = (char*)d_ws;
  bf16* wxT  = (bf16*)(ws + WS_WXT);
  bf16* whT  = (bf16*)(ws + WS_WHT);
  bf16* hbuf = (bf16*)(ws + WS_HBUF);
  int*  bar  = (int*)(ws + WS_BAR);

  // 1) convert/transpose weights, init h buffer + barrier
  rnn_prep_kernel<<<(HDIM * HDIM + 255) / 256, 256, 0, stream>>>(
      Wx, Wh, h0, wxT, whT, hbuf, bar);

  // 2) x projection GEMM -> d_out in [L,B,H] layout
  rnn_xproj_kernel<<<(SEQL * BATCH) / 16, 256, 0, stream>>>(x, wxT, bx, out);

  // 3) persistent recurrence over 1024 steps
  rnn_scan_kernel<<<NBLK_RNN, 512, 0, stream>>>(whT, bh, hbuf, out, bar);
}
